// Attention_56281251446850
// MI455X (gfx1250) — matrix operand order repaired and checked by reading, not compiled, not run on hardware
//
#include <hip/hip_runtime.h>

// ---------------------------------------------------------------------------
// Multi-head attention for MI455X (gfx1250), BF16 WMMA pipeline with
// double-buffered LDS staging for both GEMMs and the attention K/V stream.
// B=8, N=1024, DIM=1024, HEADS=16, DIM_HEAD=64, INNER=1024
// ---------------------------------------------------------------------------

#define B_      8
#define N_      1024
#define DIM_    1024
#define HEADS_  16
#define DH_     64
#define INNER_  1024
#define TOK_    (B_ * N_)          // 8192 tokens

typedef __bf16 bf16;
typedef __attribute__((ext_vector_type(16))) __bf16 v16bf;
typedef __attribute__((ext_vector_type(8)))  __bf16 v8bf;
typedef __attribute__((ext_vector_type(8)))  float  v8f;

union V16U { v16bf v; v8bf h[2]; };

// D = A(16x32 bf16) * B(32x16 bf16) + C(16x16 f32)
__device__ __forceinline__ v8f wmma_bf16(v16bf a, v16bf b, v8f c) {
  return __builtin_amdgcn_wmma_f32_16x16x32_bf16(
      /*neg_a=*/false, a, /*neg_b=*/false, b,
      /*c_mod=*/(short)0, c, /*reuse_a=*/false, /*reuse_b=*/false);
}

// A-fragment (16x32, M x K) from row-major [M][ld] global bf16.
// Contiguous k order, same as B frags: lane L holds row M=L%16; e=0..15 -> K=(L<16?0:16)+e.
__device__ __forceinline__ v16bf load_a_frag_g(const bf16* __restrict__ base,
                                               int ld, int m0, int k0, int lane) {
  const int row = m0 + (lane & 15);
  const int kk  = k0 + ((lane < 16) ? 0 : 16);
  V16U u;
  u.h[0] = *(const v8bf*)(base + (size_t)row * ld + kk);
  u.h[1] = *(const v8bf*)(base + (size_t)row * ld + kk + 8);
  return u.v;
}

// ---------------------------------------------------------------------------
// Kernel 1: f32 -> bf16 elementwise convert
// ---------------------------------------------------------------------------
__global__ void k_f32_to_bf16(const float* __restrict__ src,
                              bf16* __restrict__ dst, int n) {
  int i = blockIdx.x * blockDim.x + threadIdx.x;
  if (i < n) dst[i] = (bf16)src[i];
}

// ---------------------------------------------------------------------------
// Kernel 2: transpose + convert. src[rows][cols] f32 -> dst[cols][rows] bf16
// ---------------------------------------------------------------------------
__global__ void k_transpose_bf16(const float* __restrict__ src,
                                 bf16* __restrict__ dst, int rows, int cols) {
  int i = blockIdx.x * blockDim.x + threadIdx.x;
  if (i < rows * cols) {
    int r = i / cols, c = i % cols;
    dst[(size_t)c * rows + r] = (bf16)src[i];
  }
}

// ---------------------------------------------------------------------------
// Kernel 3: QKV GEMM.  [TOK, DIM] x [DIM, 3*INNER] -> Q,K (row major per head),
// Vt (transposed per head: [b][h][dh][n]).
// Block = 256 thr = 8 waves as 2(M) x 4(N); block tile 64x128, wave tile 32x32.
// Double-buffered LDS staging of A(64x32) and B(128x32) k-slabs.
// ---------------------------------------------------------------------------
__global__ void k_gemm_qkv(const bf16* __restrict__ X,
                           const bf16* __restrict__ Wt,   // [3*INNER][DIM]
                           bf16* __restrict__ Q,
                           bf16* __restrict__ K,
                           bf16* __restrict__ Vt) {
  __shared__ bf16 sA[2][64][32];    // 8 KB
  __shared__ bf16 sB[2][128][32];   // 16 KB

  const int t     = threadIdx.x;
  const int lane  = t & 31;
  const int wslot = t >> 5;
  const int mw = wslot & 1;         // 0..1
  const int nw = wslot >> 1;        // 0..3

  const int nblocks = (3 * INNER_) / 128;   // 24
  const int mblk = blockIdx.x / nblocks;
  const int nblk = blockIdx.x % nblocks;
  const int m0b = mblk * 64, n0b = nblk * 128;

  // cooperative staging coordinates
  const int arow = t >> 2, apart = (t & 3) * 8;    // A: 64 rows x 4 x 16B
  const int brow = t >> 1, bpart = (t & 1) * 16;   // B: 128 rows x 2 x 32B

  // prologue: stage k0 = 0 into buffer 0
  *(v8bf*)&sA[0][arow][apart] =
      *(const v8bf*)(X + (size_t)(m0b + arow) * DIM_ + apart);
  *(v16bf*)&sB[0][brow][bpart] =
      *(const v16bf*)(Wt + (size_t)(n0b + brow) * DIM_ + bpart);
  __syncthreads();

  v8f acc[2][2] = {};
  const int STEPS = DIM_ / 32;      // 32
  for (int ks = 0; ks < STEPS; ++ks) {
    const int buf = ks & 1;
    if (ks + 1 < STEPS) {           // stage next k-slab into other buffer
      const int k0 = (ks + 1) * 32;
      *(v8bf*)&sA[buf ^ 1][arow][apart] =
          *(const v8bf*)(X + (size_t)(m0b + arow) * DIM_ + k0 + apart);
      *(v16bf*)&sB[buf ^ 1][brow][bpart] =
          *(const v16bf*)(Wt + (size_t)(n0b + brow) * DIM_ + k0 + bpart);
    }
    // fragments from LDS
    const int akk = (lane < 16) ? 0 : 16;
    const int bkk = (lane < 16) ? 0 : 16;
    v16bf a0, a1;
    { V16U u;
      u.h[0] = *(const v8bf*)&sA[buf][mw * 32 + (lane & 15)][akk];
      u.h[1] = *(const v8bf*)&sA[buf][mw * 32 + (lane & 15)][akk + 8];
      a0 = u.v; }
    { V16U u;
      u.h[0] = *(const v8bf*)&sA[buf][mw * 32 + 16 + (lane & 15)][akk];
      u.h[1] = *(const v8bf*)&sA[buf][mw * 32 + 16 + (lane & 15)][akk + 8];
      a1 = u.v; }
    v16bf b0 = *(const v16bf*)&sB[buf][nw * 32 + (lane & 15)][bkk];
    v16bf b1 = *(const v16bf*)&sB[buf][nw * 32 + 16 + (lane & 15)][bkk];

    acc[0][0] = wmma_bf16(a0, b0, acc[0][0]);
    acc[0][1] = wmma_bf16(a0, b1, acc[0][1]);
    acc[1][0] = wmma_bf16(a1, b0, acc[1][0]);
    acc[1][1] = wmma_bf16(a1, b1, acc[1][1]);
    __syncthreads();
  }

  // epilogue: scatter into Q / K / Vt (bf16)
  const int m0 = m0b + mw * 32;
  const int n0 = n0b + nw * 32;
  const int rbase = (lane < 16) ? 0 : 8;
  #pragma unroll
  for (int im = 0; im < 2; ++im) {
    #pragma unroll
    for (int in = 0; in < 2; ++in) {
      const int col   = n0 + in * 16 + (lane & 15);
      const int sel   = col >> 10;           // 0=Q, 1=K, 2=V
      const int inner = col & (INNER_ - 1);
      const int h     = inner >> 6;
      const int dh    = inner & (DH_ - 1);
      #pragma unroll
      for (int r = 0; r < 8; ++r) {
        const int row = m0 + im * 16 + rbase + r;
        const int b   = row >> 10;
        const int n   = row & (N_ - 1);
        const size_t bh = (size_t)b * HEADS_ + h;
        const bf16 v = (bf16)acc[im][in][r];
        if (sel == 0)      Q[(bh * N_ + n) * DH_ + dh]  = v;
        else if (sel == 1) K[(bh * N_ + n) * DH_ + dh]  = v;
        else               Vt[(bh * DH_ + dh) * N_ + n] = v;
      }
    }
  }
}

// ---------------------------------------------------------------------------
// Kernel 4: flash-style attention, workgroup-cooperative K/V staging.
// One block (8 waves) owns one (b, h, 128-query slab); each wave owns a
// 16-query tile. Keys streamed 32 at a time through double-buffered LDS
// (K tile 32x64, V tile 64x32), loaded ONCE per block instead of per wave.
// ---------------------------------------------------------------------------
__global__ void k_attention(const bf16* __restrict__ Q,
                            const bf16* __restrict__ K,
                            const bf16* __restrict__ Vt,
                            bf16* __restrict__ O) {
  __shared__ bf16 sK[2][32][64];     // 8 KB  (keys x dh)
  __shared__ bf16 sV[2][64][32];     // 8 KB  (dh x keys, from Vt)
  __shared__ bf16 sP[8][16][32];     // 8 KB  per-wave P tiles

  const int t     = threadIdx.x;
  const int lane  = t & 31;
  const int wslot = t >> 5;

  const int bh   = blockIdx.x >> 3;          // 0..127
  const int qblk = blockIdx.x & 7;
  const int b = bh >> 4, h = bh & 15;
  const int m0 = qblk * 128 + wslot * 16;

  const bf16* Qb = Q  + (size_t)bh * N_ * DH_;
  const bf16* Kb = K  + (size_t)bh * N_ * DH_;
  const bf16* Vb = Vt + (size_t)bh * DH_ * N_;
  const float scale = 0.125f;                // DIM_HEAD^-0.5

  // Q tile A-fragments (read once, straight from global)
  const v16bf qa0 = load_a_frag_g(Qb, DH_, m0, 0,  lane);
  const v16bf qa1 = load_a_frag_g(Qb, DH_, m0, 32, lane);

  // cooperative staging coordinates
  const int vrow = t >> 2, vpart = (t & 3) * 8;   // V: 64 rows x 4 x 16B

  // prologue: stage keys [0,32) into buffer 0
  *(v8bf*)((bf16*)&sK[0][0][0] + t * 8) = *(const v8bf*)(Kb + t * 8);
  *(v8bf*)&sV[0][vrow][vpart] = *(const v8bf*)(Vb + (size_t)vrow * N_ + vpart);
  __syncthreads();

  v8f acc[4] = {};                   // 16x64 output accumulator (4 C tiles)
  float mrow[8], lrow[8];
  #pragma unroll
  for (int r = 0; r < 8; ++r) { mrow[r] = -1e30f; lrow[r] = 0.0f; }

  const int rbase = (lane < 16) ? 0 : 8;
  const int STEPS = N_ / 32;         // 32
  for (int it = 0; it < STEPS; ++it) {
    const int buf = it & 1;
    if (it + 1 < STEPS) {            // stage next 32 keys into other buffer
      const int j0n = (it + 1) * 32;
      *(v8bf*)((bf16*)&sK[buf ^ 1][0][0] + t * 8) =
          *(const v8bf*)(Kb + (size_t)j0n * DH_ + t * 8);
      *(v8bf*)&sV[buf ^ 1][vrow][vpart] =
          *(const v8bf*)(Vb + (size_t)vrow * N_ + j0n + vpart);
    }

    // --- S = scale * Q K^T for 32 staged keys (two 16x16 tiles) ---
    const int bkk = (lane < 16) ? 0 : 16;
    v16bf kb00 = *(const v16bf*)&sK[buf][(lane & 15)][bkk];
    v16bf kb01 = *(const v16bf*)&sK[buf][(lane & 15)][32 + bkk];
    v16bf kb10 = *(const v16bf*)&sK[buf][16 + (lane & 15)][bkk];
    v16bf kb11 = *(const v16bf*)&sK[buf][16 + (lane & 15)][32 + bkk];
    v8f s0 = {}, s1 = {};
    s0 = wmma_bf16(qa0, kb00, s0);
    s0 = wmma_bf16(qa1, kb01, s0);
    s1 = wmma_bf16(qa0, kb10, s1);
    s1 = wmma_bf16(qa1, kb11, s1);

    // --- online softmax over the 32 new keys ---
    float p0[8], p1[8];
    #pragma unroll
    for (int r = 0; r < 8; ++r) {
      float a = s0[r] * scale;
      float c = s1[r] * scale;
      float v = fmaxf(a, c);         // row max lives across a 16-lane half
      v = fmaxf(v, __shfl_xor(v, 1, 16));
      v = fmaxf(v, __shfl_xor(v, 2, 16));
      v = fmaxf(v, __shfl_xor(v, 4, 16));
      v = fmaxf(v, __shfl_xor(v, 8, 16));
      const float mnew  = fmaxf(mrow[r], v);
      const float alpha = __expf(mrow[r] - mnew);
      mrow[r] = mnew;
      p0[r] = __expf(a - mnew);
      p1[r] = __expf(c - mnew);
      float rs = p0[r] + p1[r];
      rs += __shfl_xor(rs, 1, 16);
      rs += __shfl_xor(rs, 2, 16);
      rs += __shfl_xor(rs, 4, 16);
      rs += __shfl_xor(rs, 8, 16);
      lrow[r] = lrow[r] * alpha + rs;
      #pragma unroll
      for (int d = 0; d < 4; ++d) acc[d][r] *= alpha;   // rescale O
    }

    // --- P (C-layout) -> per-wave LDS tile (row-major 16x32 bf16) ---
    const int kcol = lane & 15;
    #pragma unroll
    for (int r = 0; r < 8; ++r) {
      sP[wslot][rbase + r][kcol]      = (bf16)p0[r];
      sP[wslot][rbase + r][16 + kcol] = (bf16)p1[r];
    }
    // same-wave DS ops are in-order; compiler inserts the dscnt waits

    // --- P A-fragment from LDS ---
    V16U pu;
    const int prow = lane & 15;
    const int pk   = (lane < 16) ? 0 : 16;
    pu.h[0] = *(const v8bf*)&sP[wslot][prow][pk];
    pu.h[1] = *(const v8bf*)&sP[wslot][prow][pk + 8];

    // --- O += P * V from staged V tile ---
    #pragma unroll
    for (int d = 0; d < 4; ++d) {
      v16bf vb = *(const v16bf*)&sV[buf][d * 16 + (lane & 15)][bkk];
      acc[d] = wmma_bf16(pu.v, vb, acc[d]);
    }
    __syncthreads();                 // staged buf^1 ready; buf free to reuse
  }

  // --- normalize and write O (merged heads, row-major [TOK][INNER]) ---
  #pragma unroll
  for (int r = 0; r < 8; ++r) lrow[r] = 1.0f / lrow[r];
  #pragma unroll
  for (int d = 0; d < 4; ++d) {
    #pragma unroll
    for (int r = 0; r < 8; ++r) {
      const int row = m0 + rbase + r;
      const int col = h * DH_ + d * 16 + (lane & 15);
      O[((size_t)b * N_ + row) * INNER_ + col] = (bf16)(acc[d][r] * lrow[r]);
    }
  }
}

// ---------------------------------------------------------------------------
// Kernel 5: output projection. [TOK, INNER] x [INNER, DIM] + bias -> f32 out.
// Same 64x128 block tiling with double-buffered LDS staging.
// ---------------------------------------------------------------------------
__global__ void k_gemm_out(const bf16* __restrict__ O,
                           const bf16* __restrict__ WoT,   // [DIM][INNER]
                           const float* __restrict__ bias,
                           float* __restrict__ Y) {
  __shared__ bf16 sA[2][64][32];
  __shared__ bf16 sB[2][128][32];

  const int t     = threadIdx.x;
  const int lane  = t & 31;
  const int wslot = t >> 5;
  const int mw = wslot & 1;
  const int nw = wslot >> 1;

  const int nblocks = DIM_ / 128;            // 8
  const int mblk = blockIdx.x / nblocks;
  const int nblk = blockIdx.x % nblocks;
  const int m0b = mblk * 64, n0b = nblk * 128;

  const int arow = t >> 2, apart = (t & 3) * 8;
  const int brow = t >> 1, bpart = (t & 1) * 16;

  *(v8bf*)&sA[0][arow][apart] =
      *(const v8bf*)(O + (size_t)(m0b + arow) * INNER_ + apart);
  *(v16bf*)&sB[0][brow][bpart] =
      *(const v16bf*)(WoT + (size_t)(n0b + brow) * INNER_ + bpart);
  __syncthreads();

  v8f acc[2][2] = {};
  const int STEPS = INNER_ / 32;
  for (int ks = 0; ks < STEPS; ++ks) {
    const int buf = ks & 1;
    if (ks + 1 < STEPS) {
      const int k0 = (ks + 1) * 32;
      *(v8bf*)&sA[buf ^ 1][arow][apart] =
          *(const v8bf*)(O + (size_t)(m0b + arow) * INNER_ + k0 + apart);
      *(v16bf*)&sB[buf ^ 1][brow][bpart] =
          *(const v16bf*)(WoT + (size_t)(n0b + brow) * INNER_ + k0 + bpart);
    }
    const int akk = (lane < 16) ? 0 : 16;
    const int bkk = (lane < 16) ? 0 : 16;
    v16bf a0, a1;
    { V16U u;
      u.h[0] = *(const v8bf*)&sA[buf][mw * 32 + (lane & 15)][akk];
      u.h[1] = *(const v8bf*)&sA[buf][mw * 32 + (lane & 15)][akk + 8];
      a0 = u.v; }
    { V16U u;
      u.h[0] = *(const v8bf*)&sA[buf][mw * 32 + 16 + (lane & 15)][akk];
      u.h[1] = *(const v8bf*)&sA[buf][mw * 32 + 16 + (lane & 15)][akk + 8];
      a1 = u.v; }
    v16bf b0 = *(const v16bf*)&sB[buf][nw * 32 + (lane & 15)][bkk];
    v16bf b1 = *(const v16bf*)&sB[buf][nw * 32 + 16 + (lane & 15)][bkk];

    acc[0][0] = wmma_bf16(a0, b0, acc[0][0]);
    acc[0][1] = wmma_bf16(a0, b1, acc[0][1]);
    acc[1][0] = wmma_bf16(a1, b0, acc[1][0]);
    acc[1][1] = wmma_bf16(a1, b1, acc[1][1]);
    __syncthreads();
  }

  const int m0 = m0b + mw * 32;
  const int n0 = n0b + nw * 32;
  const int rbase = (lane < 16) ? 0 : 8;
  #pragma unroll
  for (int im = 0; im < 2; ++im) {
    #pragma unroll
    for (int in = 0; in < 2; ++in) {
      const int col = n0 + in * 16 + (lane & 15);
      const float bv = bias[col];
      #pragma unroll
      for (int r = 0; r < 8; ++r) {
        const int row = m0 + im * 16 + rbase + r;
        Y[(size_t)row * DIM_ + col] = acc[im][in][r] + bv;
      }
    }
  }
}

// ---------------------------------------------------------------------------
// Host-side launch
// ---------------------------------------------------------------------------
extern "C" void kernel_launch(void* const* d_in, const int* in_sizes, int n_in,
                              void* d_out, int out_size, void* d_ws, size_t ws_size,
                              hipStream_t stream) {
  const float* x     = (const float*)d_in[0];   // [8, 1024, 1024]
  const float* w_qkv = (const float*)d_in[1];   // [1024, 3072]
  const float* w_out = (const float*)d_in[2];   // [1024, 1024]
  const float* b_out = (const float*)d_in[3];   // [1024]
  float* y = (float*)d_out;                     // [8, 1024, 1024]

  char* ws = (char*)d_ws;
  const size_t MB = 1024 * 1024;
  bf16* Xh     = (bf16*)(ws + 0);          // 16 MiB : [TOK][DIM]
  bf16* Wqkvt  = (bf16*)(ws + 16 * MB);    //  6 MiB : [3*INNER][DIM]
  bf16* Wot    = (bf16*)(ws + 22 * MB);    //  2 MiB : [DIM][INNER]
  bf16* Qm     = (bf16*)(ws + 24 * MB);    // 16 MiB : [B][H][N][Dh]
  bf16* Km     = (bf16*)(ws + 40 * MB);    // 16 MiB : [B][H][N][Dh]
  bf16* Vtm    = (bf16*)(ws + 56 * MB);    // 16 MiB : [B][H][Dh][N]
  bf16* Om     = (bf16*)(ws + 72 * MB);    // 16 MiB : [TOK][INNER]

  // 1) convert X
  {
    const int n = TOK_ * DIM_;
    k_f32_to_bf16<<<n / 256, 256, 0, stream>>>(x, Xh, n);
  }
  // 2) transpose + convert weights
  {
    const int n1 = DIM_ * 3 * INNER_;
    k_transpose_bf16<<<(n1 + 255) / 256, 256, 0, stream>>>(w_qkv, Wqkvt, DIM_, 3 * INNER_);
    const int n2 = INNER_ * DIM_;
    k_transpose_bf16<<<(n2 + 255) / 256, 256, 0, stream>>>(w_out, Wot, INNER_, DIM_);
  }
  // 3) QKV GEMM: (TOK/64) x (3072/128) = 128 x 24 = 3072 blocks
  k_gemm_qkv<<<3072, 256, 0, stream>>>(Xh, Wqkvt, Qm, Km, Vtm);
  // 4) attention: (B*H) x (N/128) = 128 x 8 = 1024 blocks
  k_attention<<<1024, 256, 0, stream>>>(Qm, Km, Vtm, Om);
  // 5) output projection: (TOK/64) x (DIM/128) = 128 x 8 = 1024 blocks
  k_gemm_out<<<1024, 256, 0, stream>>>(Om, Wot, b_out, y);
}
